// SparseKeypointDetector_4277787427005
// MI455X (gfx1250) — compile-verified, matrix operand order checked by reading, not run on hardware
//
#include <hip/hip_runtime.h>
#include <hip/hip_bf16.h>

typedef __attribute__((ext_vector_type(16))) _Float16 v16h;
typedef __attribute__((ext_vector_type(8)))  float    v8f;
typedef __attribute__((ext_vector_type(8)))  unsigned int v8u;
typedef unsigned int u32;
typedef unsigned long long u64;

constexpr int kB = 8, kH = 1024, kW = 2048;
constexpr int kHW = kH * kW;                 // 2^21 per image
constexpr int kBins = 4096;                  // linear-value histogram bins
constexpr int kCap = 4096;                   // candidate buffer capacity per image
constexpr int kTopK = 2048;                  // reference TOPK
constexpr int kMaxKP = 512;
constexpr int kBlkPerImg = 32;
constexpr int kChunk = kHW / kBlkPerImg;     // 65536 floats per block
constexpr float kThresh = 0.1f;
constexpr float kR2 = 9.0f;                  // NMS radius^2

// half <-> packed helpers (coords <= 2047 are exact in f16)
__device__ inline u32 pack_xy(float x, float y) {
    unsigned short hx = __builtin_bit_cast(unsigned short, (_Float16)x);
    unsigned short hy = __builtin_bit_cast(unsigned short, (_Float16)y);
    return (u32)hx | ((u32)hy << 16);
}
__device__ inline float unpack_x(u32 p) {
    return (float)__builtin_bit_cast(_Float16, (unsigned short)(p & 0xffffu));
}
__device__ inline float unpack_y(u32 p) {
    return (float)__builtin_bit_cast(_Float16, (unsigned short)(p >> 16));
}

// ---------------- ws layout ----------------
// u32 ghist[B][kBins]  : 128 KB
// u32 gcnt[B], gcut[B] : 64 B
// u64 gcand[B][kCap]   : 256 KB

__global__ __launch_bounds__(256) void init_ws_kernel(u32* p, int n) {
    int i = blockIdx.x * 256 + threadIdx.x;
    if (i < n) p[i] = 0;
}

// Pass 1: per-image histogram of v = img*mask on a linear [0,1) scale.
__global__ __launch_bounds__(256) void hist_kernel(const float* __restrict__ img,
                                                   const float* __restrict__ msk,
                                                   u32* __restrict__ ghist) {
    __shared__ u32 lh[kBins];
    const int tid = threadIdx.x;
    for (int i = tid; i < kBins; i += 256) lh[i] = 0;
    __syncthreads();
    const int im  = blockIdx.x / kBlkPerImg;
    const int blk = blockIdx.x % kBlkPerImg;
    const float4* ip = (const float4*)img + (size_t)im * (kHW / 4) + (size_t)blk * (kChunk / 4);
    const float4* mp = (const float4*)msk + (size_t)im * (kHW / 4) + (size_t)blk * (kChunk / 4);
    for (int k = 0; k < kChunk / 4 / 256; ++k) {
        int q = k * 256 + tid;
        float4 a = ip[q];
        float4 m = mp[q];
        float v[4] = {a.x * m.x, a.y * m.y, a.z * m.z, a.w * m.w};
#pragma unroll
        for (int c = 0; c < 4; ++c) {
            u32 bn = (u32)(v[c] * (float)kBins);
            if (bn > (u32)(kBins - 1)) bn = kBins - 1;
            atomicAdd(&lh[bn], 1u);
        }
    }
    __syncthreads();
    for (int i = tid; i < kBins; i += 256) {
        u32 h = lh[i];
        if (h) atomicAdd(&ghist[im * kBins + i], h);
    }
}

// Pass 2: find cut bin such that count(bin >= cut) >= kTopK.
__global__ void select_kernel(const u32* __restrict__ ghist, u32* __restrict__ gcut) {
    if (threadIdx.x == 0) {
        int im = blockIdx.x;
        u32 acc = 0;
        int bin = 0;
        for (int i = kBins - 1; i >= 0; --i) {
            acc += ghist[im * kBins + i];
            if (acc >= (u32)kTopK) { bin = i; break; }
        }
        gcut[im] = (u32)bin;
    }
}

// Pass 3: gather candidates with bin >= cut (re-read hits the 192MB L2).
// Key = {value bits : 32 | ~index : 32}: descending sort -> value desc,
// ties by ascending index (top_k tie order).
__global__ __launch_bounds__(256) void compact_kernel(const float* __restrict__ img,
                                                      const float* __restrict__ msk,
                                                      const u32* __restrict__ gcut,
                                                      u32* __restrict__ gcnt,
                                                      u64* __restrict__ gcand) {
    const int tid = threadIdx.x;
    const int im  = blockIdx.x / kBlkPerImg;
    const int blk = blockIdx.x % kBlkPerImg;
    const u32 cut = gcut[im];
    const float4* ip = (const float4*)img + (size_t)im * (kHW / 4) + (size_t)blk * (kChunk / 4);
    const float4* mp = (const float4*)msk + (size_t)im * (kHW / 4) + (size_t)blk * (kChunk / 4);
    for (int k = 0; k < kChunk / 4 / 256; ++k) {
        int q = k * 256 + tid;
        float4 a = ip[q];
        float4 m = mp[q];
        float v[4] = {a.x * m.x, a.y * m.y, a.z * m.z, a.w * m.w};
#pragma unroll
        for (int c = 0; c < 4; ++c) {
            u32 bn = (u32)(v[c] * (float)kBins);
            if (bn > (u32)(kBins - 1)) bn = kBins - 1;
            if (bn >= cut) {
                u32 pos = atomicAdd(&gcnt[im], 1u);
                if (pos < (u32)kCap) {
                    u32 e = (u32)(blk * kChunk + q * 4 + c);
                    gcand[(size_t)im * kCap + pos] =
                        ((u64)__float_as_uint(v[c]) << 32) | (u64)(0xFFFFFFFFu - e);
                }
            }
        }
    }
}

// Pass 4 (one block per image): bitonic sort desc, greedy NMS with WMMA-tiled
// pairwise distances, compact survivors.
__global__ __launch_bounds__(256) void finalize_kernel(const u32* __restrict__ gcnt,
                                                       const u64* __restrict__ gcand,
                                                       float* __restrict__ out) {
    __shared__ u64 keys[kCap];              // 32 KB
    __shared__ u32 xyp[kTopK];              // 8 KB packed (half x, half y)
    __shared__ float vals[kTopK];           // 8 KB
    __shared__ float norms[kTopK];          // 8 KB
    __shared__ u32 keepw[kTopK / 32];
    __shared__ u32 wcnt[kTopK / 32];
    __shared__ u32 woff[kTopK / 32];

    const int tid = threadIdx.x;
    const int im  = blockIdx.x;

    // ---- load candidates, pad with 0 (sinks to end under descending sort)
    u32 n = gcnt[im];
    if (n > (u32)kCap) n = kCap;
    for (int c = tid; c < kCap; c += 256)
        keys[c] = (c < (int)n) ? gcand[(size_t)im * kCap + c] : 0ull;

    // ---- bitonic sort, descending
    for (u32 k = 2; k <= (u32)kCap; k <<= 1) {
        for (u32 j = k >> 1; j > 0; j >>= 1) {
            __syncthreads();
            for (u32 t = tid; t < (u32)kCap / 2; t += 256) {
                u32 i1 = (t << 1) - (t & (j - 1));
                u32 i2 = i1 + j;
                bool desc = ((i1 & k) == 0);
                u64 A = keys[i1], Bv = keys[i2];
                bool sw = desc ? (A < Bv) : (A > Bv);
                if (sw) { keys[i1] = Bv; keys[i2] = A; }
            }
        }
    }
    __syncthreads();

    // ---- unpack top 2048: packed coords, values, norms
    for (int c = tid; c < kTopK; c += 256) {
        u64 key = keys[c];
        u32 vb = (u32)(key >> 32);
        u32 e  = 0xFFFFFFFFu - (u32)(key & 0xFFFFFFFFull);
        float v = __uint_as_float(vb);
        float x = (float)(e & (u32)(kW - 1));   // e % 2048
        float y = (float)(e >> 11);             // e / 2048
        vals[c]  = v;
        xyp[c]   = pack_xy(x, y);
        norms[c] = x * x + y * y;
    }
    __syncthreads();

    // ---- initial keep = (val > thresh)
    if (tid < kTopK / 32) {
        u32 w = 0;
        for (int b = 0; b < 32; ++b)
            if (vals[tid * 32 + b] > kThresh) w |= (1u << b);
        keepw[tid] = w;
    }
    __syncthreads();

    // ---- greedy NMS in 16-row blocks; WMMA tiles suppress later columns.
    // Lanes l and l+16 load the same (x,y) word: the duplicated K=0,1 / K=8,9
    // slots make the WMMA produce d[r] = 2*(x_m x_n + y_m y_n) directly.
    const int lane = tid & 31;
    const int wave = tid >> 5;
    const float kInf = __builtin_inff();
    for (int b = 0; b < kTopK / 16; ++b) {
        const int base = b * 16;
        if (tid == 0) {
            // sequential greedy inside the 16-row block (exclusive phase)
            for (int i = 0; i < 15; ++i) {
                int gi = base + i;
                if ((keepw[gi >> 5] >> (gi & 31)) & 1) {
                    u32 pi = xyp[gi];
                    float xi = unpack_x(pi), yi = unpack_y(pi);
                    for (int jj = i + 1; jj < 16; ++jj) {
                        int gj = base + jj;
                        if ((keepw[gj >> 5] >> (gj & 31)) & 1) {
                            u32 pj = xyp[gj];
                            float dx = xi - unpack_x(pj);
                            float dy = yi - unpack_y(pj);
                            if (dx * dx + dy * dy < kR2)
                                keepw[gj >> 5] &= ~(1u << (gj & 31));
                        }
                    }
                }
            }
        }
        __syncthreads();

        // A operand: unconditional load, row (lane&15) at K=0,1 and K=8,9
        v8u au = {};
        au[0] = xyp[base + (lane & 15)];
        const v16h a = __builtin_bit_cast(v16h, au);

        // block-invariant row state: keep bit folded into row norm (+INF kills)
        const u32 rowWord = keepw[base >> 5];
        const int rshift = base & 31;
        const int mbase = (lane < 16) ? 0 : 8;
        const int nn = lane & 15;
        float rn[8];
#pragma unroll
        for (int r = 0; r < 8; ++r) {
            bool kept = (rowWord >> (rshift + mbase + r)) & 1;
            rn[r] = kept ? norms[base + mbase + r] : kInf;
        }

        for (int t = b + 1 + wave; t < kTopK / 16; t += 8) {
            v8u bu = {};
            bu[0] = xyp[t * 16 + (lane & 15)];
            v16h bb = __builtin_bit_cast(v16h, bu);
            v8f cz = {};
            v8f d = __builtin_amdgcn_wmma_f32_16x16x32_f16(
                false, a, false, bb, (short)0, cz, false, false);
            int j = t * 16 + nn;
            float nj = norms[j];
            // mn = min_r (rn[r] - 2*dot(r,n)); suppress if mn + nj < R^2
            float m01 = fminf(rn[0] - d[0], rn[1] - d[1]);
            float m23 = fminf(rn[2] - d[2], rn[3] - d[3]);
            float m45 = fminf(rn[4] - d[4], rn[5] - d[5]);
            float m67 = fminf(rn[6] - d[6], rn[7] - d[7]);
            float mn = fminf(fminf(m01, m23), fminf(m45, m67));
            if (mn + nj < kR2) atomicAnd(&keepw[j >> 5], ~(1u << (j & 31)));
        }
        __syncthreads();
    }

    // ---- compact survivors (score order), zero-pad to MAX_KP
    for (int i = tid; i < kMaxKP * 2; i += 256) out[im * kMaxKP * 2 + i] = 0.0f;
    for (int i = tid; i < kMaxKP; i += 256) out[kB * kMaxKP * 2 + im * kMaxKP + i] = 0.0f;
    __syncthreads();
    if (tid < kTopK / 32) wcnt[tid] = __popc(keepw[tid]);
    __syncthreads();
    if (tid == 0) {
        u32 acc = 0;
        for (int w = 0; w < kTopK / 32; ++w) { woff[w] = acc; acc += wcnt[w]; }
    }
    __syncthreads();
    if (tid < kTopK / 32) {
        u32 w = keepw[tid];
        u32 pos = woff[tid];
        while (w) {
            int bbit = __ffs(w) - 1;
            w &= (w - 1);
            if (pos < (u32)kMaxKP) {
                int c = tid * 32 + bbit;
                u32 p = xyp[c];
                out[im * kMaxKP * 2 + pos * 2 + 0] = unpack_x(p);
                out[im * kMaxKP * 2 + pos * 2 + 1] = unpack_y(p);
                out[kB * kMaxKP * 2 + im * kMaxKP + pos] = vals[c];
            }
            pos++;
        }
    }
}

extern "C" void kernel_launch(void* const* d_in, const int* in_sizes, int n_in,
                              void* d_out, int out_size, void* d_ws, size_t ws_size,
                              hipStream_t stream) {
    const float* img = (const float*)d_in[0];
    const float* msk = (const float*)d_in[1];
    float* out = (float*)d_out;

    u32* ghist = (u32*)d_ws;                         // B*kBins
    u32* gcnt  = ghist + kB * kBins;                 // B
    u32* gcut  = gcnt + kB;                          // B
    u64* gcand = (u64*)((char*)d_ws + (size_t)(kB * kBins + 2 * kB) * sizeof(u32));

    const int nz = kB * kBins + kB;                  // hist + counters (contiguous)
    init_ws_kernel<<<(nz + 255) / 256, 256, 0, stream>>>(ghist, nz);
    hist_kernel<<<kB * kBlkPerImg, 256, 0, stream>>>(img, msk, ghist);
    select_kernel<<<kB, 32, 0, stream>>>(ghist, gcut);
    compact_kernel<<<kB * kBlkPerImg, 256, 0, stream>>>(img, msk, gcut, gcnt, gcand);
    finalize_kernel<<<kB, 256, 0, stream>>>(gcnt, gcand, out);
}
